// BackupOpticalFC_28999619182924
// MI455X (gfx1250) — compile-verified, matrix operand order checked by reading, not run on hardware
//
#include <hip/hip_runtime.h>

// ---------------------------------------------------------------------------
// out[B,O] = x[B,I] @ W[O,I]^T + b   (fp32), B=8192, I=O=1024
// bf16x3 split-precision GEMM on CDNA5 WMMA (v_wmma_f32_16x16x32_bf16).
//
// Path A (ws large enough): one-shot split of x/W into bf16 hi/lo planes in
//   d_ws, then a pure-WMMA GEMM. Tiles staged with CDNA5 async global->LDS
//   copies (GLOBAL_LOAD_ASYNC_TO_LDS_B128, ASYNCcnt) - no VGPR transit,
//   no spill pressure.
// Path B (fallback): fused kernel that splits inside the loop.
// ---------------------------------------------------------------------------

typedef __attribute__((ext_vector_type(16))) __bf16 v16bf;
typedef __attribute__((ext_vector_type(8)))  float  v8f;
typedef __attribute__((ext_vector_type(4)))  int    i32x4;
typedef unsigned short ushort_t;

#define AS1 __attribute__((address_space(1)))
#define AS3 __attribute__((address_space(3)))

#if __has_builtin(__builtin_amdgcn_global_load_async_to_lds_b128)
#define HAVE_ASYNC_LDS 1
#else
#define HAVE_ASYNC_LDS 0
#endif

__device__ inline void wait_async_zero() {
#if __has_builtin(__builtin_amdgcn_s_wait_asynccnt)
    __builtin_amdgcn_s_wait_asynccnt(0);
#else
    asm volatile("s_wait_asynccnt 0x0" ::: "memory");
#endif
}

#define BATCH 8192
#define IN_F  1024
#define OUT_F 1024

constexpr int BM = 64;          // rows of x per block
constexpr int BN = 64;          // rows of W (cols of out) per block
constexpr int BK = 32;          // K per stage (one WMMA k-step)
constexpr int PAD = 8;          // LDS padding (elements) to break bank conflicts
constexpr int LDK = BK + PAD;   // 40 ushorts -> 80B row stride (16B aligned)
constexpr int THREADS = 128;    // 4 waves (wave32)
constexpr int STAGES = IN_F / BK;

union FragU {
    v16bf v;
    uint4 q[2];
};

// Round-to-nearest-even fp32 -> bf16 bits, plus residual split.
__device__ inline void split_bf16(float f, unsigned short& hi, unsigned short& lo) {
    unsigned int u  = __float_as_uint(f);
    unsigned int hr = (u + 0x7FFFu + ((u >> 16) & 1u)) >> 16;
    hi = (unsigned short)hr;
    float r = f - __uint_as_float(hr << 16);   // exact residual
    unsigned int ur = __float_as_uint(r);
    unsigned int lr = (ur + 0x7FFFu + ((ur >> 16) & 1u)) >> 16;
    lo = (unsigned short)lr;
}

__device__ inline void split_pack4(const float4 f, uint2& hi, uint2& lo) {
    unsigned short h0, h1, h2, h3, l0, l1, l2, l3;
    split_bf16(f.x, h0, l0);
    split_bf16(f.y, h1, l1);
    split_bf16(f.z, h2, l2);
    split_bf16(f.w, h3, l3);
    hi.x = (unsigned int)h0 | ((unsigned int)h1 << 16);
    hi.y = (unsigned int)h2 | ((unsigned int)h3 << 16);
    lo.x = (unsigned int)l0 | ((unsigned int)l1 << 16);
    lo.y = (unsigned int)l2 | ((unsigned int)l3 << 16);
}

// ---------------------------------------------------------------------------
// One-shot fp32 -> (bf16 hi, bf16 lo) plane split. Memory bound; runs once.
// ---------------------------------------------------------------------------
__global__ __launch_bounds__(256)
void split_planes_kernel(const float* __restrict__ src,
                         ushort_t* __restrict__ hi,
                         ushort_t* __restrict__ lo,
                         int n4) {
    int idx = blockIdx.x * blockDim.x + threadIdx.x;
    if (idx >= n4) return;
    float4 f = ((const float4*)src)[idx];
    uint2 h, l;
    split_pack4(f, h, l);
    ((uint2*)hi)[idx] = h;
    ((uint2*)lo)[idx] = l;
}

// ---------------------------------------------------------------------------
// Shared compute core: 12 WMMAs per k-stage, low fragment liveness
// (B frags persist across i; A frags loaded per-i).
// ---------------------------------------------------------------------------
#define DEFINE_COMPUTE(SA_HI, SA_LO, SB_HI, SB_LO)                                  \
    auto compute = [&](int buf) {                                                   \
        const int kb = khalf * 8;                                                   \
        FragU bhi[2], blo[2];                                                       \
        _Pragma("unroll")                                                           \
        for (int j = 0; j < 2; ++j) {                                               \
            const int br = wc * 32 + j * 16 + mn;                                   \
            const ushort_t* pbh = &SB_HI[buf][br][kb];                              \
            const ushort_t* pbl = &SB_LO[buf][br][kb];                              \
            bhi[j].q[0] = *(const uint4*)(pbh);                                     \
            bhi[j].q[1] = *(const uint4*)(pbh + 16);                                \
            blo[j].q[0] = *(const uint4*)(pbl);                                     \
            blo[j].q[1] = *(const uint4*)(pbl + 16);                                \
        }                                                                           \
        _Pragma("unroll")                                                           \
        for (int i = 0; i < 2; ++i) {                                               \
            const int ar = wr * 32 + i * 16 + mn;                                   \
            FragU ahi, alo;                                                         \
            const ushort_t* pah = &SA_HI[buf][ar][kb];                              \
            const ushort_t* pal = &SA_LO[buf][ar][kb];                              \
            ahi.q[0] = *(const uint4*)(pah);                                        \
            ahi.q[1] = *(const uint4*)(pah + 16);                                   \
            alo.q[0] = *(const uint4*)(pal);                                        \
            alo.q[1] = *(const uint4*)(pal + 16);                                   \
            _Pragma("unroll")                                                       \
            for (int j = 0; j < 2; ++j) {                                           \
                acc[i][j] = __builtin_amdgcn_wmma_f32_16x16x32_bf16(                \
                    false, ahi.v, false, bhi[j].v, (short)0, acc[i][j], false, false); \
                acc[i][j] = __builtin_amdgcn_wmma_f32_16x16x32_bf16(                \
                    false, ahi.v, false, blo[j].v, (short)0, acc[i][j], false, false); \
                acc[i][j] = __builtin_amdgcn_wmma_f32_16x16x32_bf16(                \
                    false, alo.v, false, bhi[j].v, (short)0, acc[i][j], false, false); \
            }                                                                       \
        }                                                                           \
    };

// ---------------------------------------------------------------------------
// Path A GEMM: inputs are pre-split bf16 planes.
// ---------------------------------------------------------------------------
__global__ __launch_bounds__(THREADS, 1)
void fc_bf16x3_gemm(const ushort_t* __restrict__ xhi,
                    const ushort_t* __restrict__ xlo,
                    const ushort_t* __restrict__ whi,
                    const ushort_t* __restrict__ wlo,
                    const float* __restrict__ bias,
                    float* __restrict__ out) {
    __shared__ ushort_t sAhi[2][BM][LDK];
    __shared__ ushort_t sAlo[2][BM][LDK];
    __shared__ ushort_t sBhi[2][BN][LDK];
    __shared__ ushort_t sBlo[2][BN][LDK];

    const int tid  = threadIdx.x;
    const int lane = tid & 31;
    const int wave = tid >> 5;
    const int wr   = wave >> 1;      // wave row (0..1) -> 32-row slab
    const int wc   = wave & 1;       // wave col (0..1) -> 32-col slab
    const int mn    = lane & 15;     // M (A) / N (B) index within 16
    const int khalf = lane >> 4;     // selects K octet pair per ISA layout

    const int rowBlock = blockIdx.y * BM;
    const int colBlock = blockIdx.x * BN;

    v8f acc[2][2];
#pragma unroll
    for (int i = 0; i < 2; ++i)
#pragma unroll
        for (int j = 0; j < 2; ++j) {
            v8f z = {0.f, 0.f, 0.f, 0.f, 0.f, 0.f, 0.f, 0.f};
            acc[i][j] = z;
        }

    DEFINE_COMPUTE(sAhi, sAlo, sBhi, sBlo)

#if HAVE_ASYNC_LDS
    // --- async HBM->LDS staging: no VGPR transit, tracked by ASYNCcnt ---
    auto issueStage = [&](int s, int buf) {
        const int k0 = s * BK;
#pragma unroll
        for (int i = 0; i < 2; ++i) {
            const int flat = i * THREADS + tid;       // 0..255
            const int r    = flat >> 2;               // 0..63
            const int c8   = (flat & 3) * 8;          // 8-element column group
            const size_t ax = (size_t)(rowBlock + r) * IN_F + k0 + c8;
            const size_t bx = (size_t)(colBlock + r) * IN_F + k0 + c8;
            __builtin_amdgcn_global_load_async_to_lds_b128(
                (AS1 i32x4*)(xhi + ax), (AS3 i32x4*)&sAhi[buf][r][c8], 0, 0);
            __builtin_amdgcn_global_load_async_to_lds_b128(
                (AS1 i32x4*)(xlo + ax), (AS3 i32x4*)&sAlo[buf][r][c8], 0, 0);
            __builtin_amdgcn_global_load_async_to_lds_b128(
                (AS1 i32x4*)(whi + bx), (AS3 i32x4*)&sBhi[buf][r][c8], 0, 0);
            __builtin_amdgcn_global_load_async_to_lds_b128(
                (AS1 i32x4*)(wlo + bx), (AS3 i32x4*)&sBlo[buf][r][c8], 0, 0);
        }
    };

    issueStage(0, 0);
    wait_async_zero();
    __syncthreads();

    for (int s = 0; s < STAGES; ++s) {
        const int buf = s & 1;
        if (s + 1 < STAGES) issueStage(s + 1, 1 - buf);  // copies run under WMMAs
        compute(buf);
        wait_async_zero();                               // this wave's copies done
        __syncthreads();                                 // all waves' copies done
    }
#else
    // --- register-staged fallback (spill-safe: only 32 staging VGPRs) ---
    uint4 aH[2], aL[2], bH[2], bL[2];

    auto loadGlobal = [&](int s) {
        const int k0 = s * BK;
#pragma unroll
        for (int i = 0; i < 2; ++i) {
            const int flat = i * THREADS + tid;
            const int r    = flat >> 2;
            const int c8   = (flat & 3) * 8;
            const size_t ax = (size_t)(rowBlock + r) * IN_F + k0 + c8;
            const size_t bx = (size_t)(colBlock + r) * IN_F + k0 + c8;
            aH[i] = *(const uint4*)(xhi + ax);
            aL[i] = *(const uint4*)(xlo + ax);
            bH[i] = *(const uint4*)(whi + bx);
            bL[i] = *(const uint4*)(wlo + bx);
        }
    };

    auto storeLds = [&](int buf) {
#pragma unroll
        for (int i = 0; i < 2; ++i) {
            const int flat = i * THREADS + tid;
            const int r    = flat >> 2;
            const int c8   = (flat & 3) * 8;
            *(uint4*)&sAhi[buf][r][c8] = aH[i];
            *(uint4*)&sAlo[buf][r][c8] = aL[i];
            *(uint4*)&sBhi[buf][r][c8] = bH[i];
            *(uint4*)&sBlo[buf][r][c8] = bL[i];
        }
    };

    loadGlobal(0);
    storeLds(0);
    __syncthreads();

    for (int s = 0; s < STAGES; ++s) {
        const int buf = s & 1;
        if (s + 1 < STAGES) loadGlobal(s + 1);
        compute(buf);
        if (s + 1 < STAGES) storeLds(1 - buf);
        __syncthreads();
    }
#endif

    // Epilogue: D layout is VGPR v -> row 8*(lane>>4)+v, col lane&15.
#pragma unroll
    for (int tm = 0; tm < 2; ++tm) {
#pragma unroll
        for (int tn = 0; tn < 2; ++tn) {
            const int col  = colBlock + wc * 32 + tn * 16 + mn;
            const float bv = bias[col];
            const int row0 = rowBlock + wr * 32 + tm * 16 + khalf * 8;
            float* op = out + (size_t)row0 * OUT_F + col;
#pragma unroll
            for (int v = 0; v < 8; ++v)
                __builtin_nontemporal_store(acc[tm][tn][v] + bv, op + (size_t)v * OUT_F);
        }
    }
}

// ---------------------------------------------------------------------------
// Path B (fallback if d_ws is too small): fused split+GEMM kernel.
// ---------------------------------------------------------------------------
__global__ __launch_bounds__(THREADS, 1)
void fc_bf16x3_fused(const float* __restrict__ x,
                     const float* __restrict__ W,
                     const float* __restrict__ bias,
                     float* __restrict__ out) {
    __shared__ ushort_t sAhi[2][BM][LDK];
    __shared__ ushort_t sAlo[2][BM][LDK];
    __shared__ ushort_t sBhi[2][BN][LDK];
    __shared__ ushort_t sBlo[2][BN][LDK];

    const int tid  = threadIdx.x;
    const int lane = tid & 31;
    const int wave = tid >> 5;
    const int wr   = wave >> 1;
    const int wc   = wave & 1;
    const int mn    = lane & 15;
    const int khalf = lane >> 4;

    const int rowBlock = blockIdx.y * BM;
    const int colBlock = blockIdx.x * BN;

    v8f acc[2][2];
#pragma unroll
    for (int i = 0; i < 2; ++i)
#pragma unroll
        for (int j = 0; j < 2; ++j) {
            v8f z = {0.f, 0.f, 0.f, 0.f, 0.f, 0.f, 0.f, 0.f};
            acc[i][j] = z;
        }

    DEFINE_COMPUTE(sAhi, sAlo, sBhi, sBlo)

    float4 aReg[4], bReg[4];

    auto loadGlobal = [&](int s) {
        const int k0 = s * BK;
#pragma unroll
        for (int i = 0; i < 4; ++i) {
            const int flat = i * THREADS + tid;
            const int r    = flat >> 3;
            const int c4   = flat & 7;
            aReg[i] = *(const float4*)(x + (size_t)(rowBlock + r) * IN_F + k0 + c4 * 4);
            bReg[i] = *(const float4*)(W + (size_t)(colBlock + r) * IN_F + k0 + c4 * 4);
        }
    };

    auto storeLds = [&](int buf) {
#pragma unroll
        for (int i = 0; i < 4; ++i) {
            const int flat = i * THREADS + tid;
            const int r    = flat >> 3;
            const int c    = (flat & 7) * 4;
            uint2 hi, lo;
            split_pack4(aReg[i], hi, lo);
            *(uint2*)&sAhi[buf][r][c] = hi;
            *(uint2*)&sAlo[buf][r][c] = lo;
            split_pack4(bReg[i], hi, lo);
            *(uint2*)&sBhi[buf][r][c] = hi;
            *(uint2*)&sBlo[buf][r][c] = lo;
        }
    };

    loadGlobal(0);
    storeLds(0);
    __syncthreads();

    for (int s = 0; s < STAGES; ++s) {
        const int buf = s & 1;
        if (s + 1 < STAGES) loadGlobal(s + 1);
        compute(buf);
        if (s + 1 < STAGES) storeLds(1 - buf);
        __syncthreads();
    }

#pragma unroll
    for (int tm = 0; tm < 2; ++tm) {
#pragma unroll
        for (int tn = 0; tn < 2; ++tn) {
            const int col  = colBlock + wc * 32 + tn * 16 + mn;
            const float bv = bias[col];
            const int row0 = rowBlock + wr * 32 + tm * 16 + khalf * 8;
            float* op = out + (size_t)row0 * OUT_F + col;
#pragma unroll
            for (int v = 0; v < 8; ++v)
                __builtin_nontemporal_store(acc[tm][tn][v] + bv, op + (size_t)v * OUT_F);
        }
    }
}

extern "C" void kernel_launch(void* const* d_in, const int* in_sizes, int n_in,
                              void* d_out, int out_size, void* d_ws, size_t ws_size,
                              hipStream_t stream) {
    const float* x = (const float*)d_in[0];
    const float* W = (const float*)d_in[1];
    const float* b = (const float*)d_in[2];
    float* out = (float*)d_out;

    const size_t xElems = (size_t)BATCH * IN_F;   // 8 Mi
    const size_t wElems = (size_t)OUT_F * IN_F;   // 1 Mi
    const size_t need   = 2 * (xElems + wElems) * sizeof(ushort_t);  // ~36 MB

    dim3 grid(OUT_F / BN, BATCH / BM);   // (16, 128)
    dim3 block(THREADS);

    if (ws_size >= need) {
        ushort_t* xhi = (ushort_t*)d_ws;
        ushort_t* xlo = xhi + xElems;
        ushort_t* whi = xlo + xElems;
        ushort_t* wlo = whi + wElems;

        const int xN4 = (int)(xElems / 4);
        const int wN4 = (int)(wElems / 4);
        hipLaunchKernelGGL(split_planes_kernel, dim3((xN4 + 255) / 256), dim3(256),
                           0, stream, x, xhi, xlo, xN4);
        hipLaunchKernelGGL(split_planes_kernel, dim3((wN4 + 255) / 256), dim3(256),
                           0, stream, W, whi, wlo, wN4);
        hipLaunchKernelGGL(fc_bf16x3_gemm, grid, block, 0, stream,
                           xhi, xlo, whi, wlo, b, out);
    } else {
        hipLaunchKernelGGL(fc_bf16x3_fused, grid, block, 0, stream, x, W, b, out);
    }
}
